// Linear_37538014167315
// MI455X (gfx1250) — matrix operand order repaired and checked by reading, not compiled, not run on hardware
//
#include <hip/hip_runtime.h>

// ---------------------------------------------------------------------------
// Block-sparse DSD matmul  y = x @ W^T  (W = nnz 32x32 fp32 tiles at block
// coords (rows, cols)).  fp32 emulated on the bf16 WMMA pipe via hi/lo split:
//   x ~= xh + xl,  w ~= wh + wl   (bf16 each)
//   x*w ~= xh*wh + xh*wl + xl*wh   (fp32 accumulate, ~2^-16 rel. error)
// Each wave owns a 32x32 output tile (2 M-tiles x 2 N-tiles -> 12 WMMA/iter).
// ---------------------------------------------------------------------------

typedef __attribute__((ext_vector_type(16))) __bf16 v16bf;
typedef __attribute__((ext_vector_type(8)))  __bf16 v8bf;
typedef __attribute__((ext_vector_type(4)))  __bf16 v4bf;
typedef __attribute__((ext_vector_type(8)))  float  v8f;

#define BATCH    2048
#define INFEAT   4096
#define OUTFEAT  4096
#define BLK      32
#define BM       256       // batch rows per workgroup (8 waves x 32 rows)
#define NTHREADS 256       // 8 wave32
#define WSTRIDE  40        // padded LDS row stride (halves) -> conflict-free b128

#define WMMA_BF16(a, b, c) \
  __builtin_amdgcn_wmma_f32_16x16x32_bf16(false, (a), false, (b), (short)0, (c), false, false)

__device__ __forceinline__ v16bf cat16(v8bf a, v8bf b) {
  return __builtin_shufflevector(a, b, 0,1,2,3,4,5,6,7,8,9,10,11,12,13,14,15);
}

__device__ __forceinline__ void bf16_split(float f, __bf16& h, __bf16& l) {
  h = (__bf16)f;
  l = (__bf16)(f - (float)h);
}

// Pre-pass: split fp32 array into two bf16 arrays (hi, lo), 4 elems/thread.
__global__ __launch_bounds__(256) void cvt_split_kernel(
    const float* __restrict__ in, unsigned short* __restrict__ hi,
    unsigned short* __restrict__ lo, int n4) {
  int i = blockIdx.x * blockDim.x + threadIdx.x;
  if (i >= n4) return;
  float4 f = ((const float4*)in)[i];
  float fv[4] = {f.x, f.y, f.z, f.w};
  v4bf h, l;
#pragma unroll
  for (int k = 0; k < 4; ++k) {
    __bf16 hv, lv;
    bf16_split(fv[k], hv, lv);
    h[k] = hv; l[k] = lv;
  }
  *(v4bf*)(hi + (size_t)i * 4) = h;
  *(v4bf*)(lo + (size_t)i * 4) = l;
}

template <bool PRE>
__global__ __launch_bounds__(NTHREADS) void bsdsd_kernel(
    const float* __restrict__ x, const float* __restrict__ w,
    const int* __restrict__ rows, const int* __restrict__ cols,
    float* __restrict__ y, int nnz,
    const unsigned short* __restrict__ xhiG, const unsigned short* __restrict__ xloG,
    const unsigned short* __restrict__ whiG, const unsigned short* __restrict__ wloG) {
  // [buf][hi/lo][32 rows x WSTRIDE halves], double-buffered
  __shared__ __attribute__((aligned(16))) __bf16 sW[2][2][32 * WSTRIDE];
  __shared__ int s_cnt;
  __shared__ int s_n[128];
  __shared__ int s_c[128];

  const int tid    = threadIdx.x;
  const int rb     = blockIdx.y;            // output row block 0..127
  const int batch0 = blockIdx.x * BM;

  // ---- build match list for this output row block (unordered scatter->list)
  if (tid == 0) s_cnt = 0;
  __syncthreads();
  for (int i = tid; i < nnz; i += NTHREADS) {
    if (rows[i] == rb) {
      int p = atomicAdd(&s_cnt, 1);   // ds_add_rtn_u32
      s_n[p] = i;
      s_c[p] = cols[i];
    }
  }
  __syncthreads();
  const int cnt = __builtin_amdgcn_readfirstlane(s_cnt);

  // ---- wave geometry (wave32): wave wv owns batch rows rowbase..rowbase+31
  const int  lane    = tid & 31;
  const int  wv      = tid >> 5;
  const int  rowbase = batch0 + wv * 32;
  const int  m       = lane & 15;
  const bool lo16    = lane < 16;
  const int  kb0     = lo16 ? 0 : 8;    // A frag: K {0..7,16..23} / {8..15,24..31}
  const int  kbB     = lo16 ? 0 : 8;    // B frag: K {0..7,16..23} / {8..15,24..31}

  // per-lane element offsets into x/xhi/xlo for the two M-tiles (col block 0);
  // per-iteration column offset (c*BLK) is added as a wave-uniform scalar.
  const int voffA0 = (rowbase + m) * INFEAT + kb0;
  const int voffA1 = (rowbase + 16 + m) * INFEAT + kb0;

  v8f acc00 = {}, acc01 = {};   // M-tile 0, N cols 0..15 / 16..31
  v8f acc10 = {}, acc11 = {};   // M-tile 1

  // ---- W staging (double buffered), per-thread register slice
  v8bf   pv  = {};
  float4 fva = {0.f, 0.f, 0.f, 0.f};
  const int sr = PRE ? (tid >> 2) : (tid >> 3);  // PRE: 0..63 (hi|lo rows), FB: 0..31
  const int sq = PRE ? (tid & 3)  : (tid & 7);

  auto stage_load = [&](int n) {
    if (PRE) {
      const unsigned short* src =
          ((sr < 32) ? whiG : wloG) + (size_t)n * 1024 + (size_t)(sr & 31) * 32 + sq * 8;
      pv = *(const v8bf*)src;                               // global_load_b128
    } else {
      fva = *(const float4*)(w + (size_t)n * 1024 + (size_t)sr * 32 + sq * 4);
    }
  };
  auto stage_store = [&](int buf) {
    if (PRE) {
      *(v8bf*)&sW[buf][sr >> 5][(sr & 31) * WSTRIDE + sq * 8] = pv;   // ds_store_b128
    } else {
      float fv[4] = {fva.x, fva.y, fva.z, fva.w};
      v4bf h, l;
#pragma unroll
      for (int k = 0; k < 4; ++k) {
        __bf16 hv, lv;
        bf16_split(fv[k], hv, lv);
        h[k] = hv; l[k] = lv;
      }
      *(v4bf*)&sW[buf][0][sr * WSTRIDE + sq * 4] = h;
      *(v4bf*)&sW[buf][1][sr * WSTRIDE + sq * 4] = l;
    }
  };

  // Load one A fragment pair (hi,lo) for M-tile with per-lane offset voffA.
  auto load_a = [&](int cc, int voffA, v16bf& ahi, v16bf& alo) {
    if (PRE) {
      const unsigned short* bh = xhiG + (size_t)cc * BLK;   // scalar base
      const unsigned short* bl = xloG + (size_t)cc * BLK;
      v8bf h0 = *(const v8bf*)(bh + voffA);
      v8bf h1 = *(const v8bf*)(bh + voffA + 16);
      v8bf l0 = *(const v8bf*)(bl + voffA);
      v8bf l1 = *(const v8bf*)(bl + voffA + 16);
      ahi = cat16(h0, h1);
      alo = cat16(l0, l1);
    } else {
      const float* xr = x + (size_t)cc * BLK + voffA;
      float4 f0 = *(const float4*)(xr);
      float4 f1 = *(const float4*)(xr + 4);
      float4 f2 = *(const float4*)(xr + 16);
      float4 f3 = *(const float4*)(xr + 20);
      float fv[16] = {f0.x, f0.y, f0.z, f0.w, f1.x, f1.y, f1.z, f1.w,
                      f2.x, f2.y, f2.z, f2.w, f3.x, f3.y, f3.z, f3.w};
#pragma unroll
      for (int i = 0; i < 16; ++i) {
        __bf16 hv, lv;
        bf16_split(fv[i], hv, lv);
        ahi[i] = hv; alo[i] = lv;
      }
    }
  };

  auto compute = [&](int buf, int cc) {
    // B fragments from LDS, B[k][o] = w[o][k]: lane reads k {0..7,16..23} / {8..15,24..31}
    // of row o = (lane&15)+16t  (2x ds_load_b128 each, conflict-free).
    const int off0 = m * WSTRIDE + kbB;
    const int off1 = off0 + 16 * WSTRIDE;
    v16bf bh0 = cat16(*(const v8bf*)&sW[buf][0][off0], *(const v8bf*)&sW[buf][0][off0 + 16]);
    v16bf bl0 = cat16(*(const v8bf*)&sW[buf][1][off0], *(const v8bf*)&sW[buf][1][off0 + 16]);
    v16bf bh1 = cat16(*(const v8bf*)&sW[buf][0][off1], *(const v8bf*)&sW[buf][0][off1 + 16]);
    v16bf bl1 = cat16(*(const v8bf*)&sW[buf][1][off1], *(const v8bf*)&sW[buf][1][off1 + 16]);

    v16bf ahi0, alo0, ahi1, alo1;
    load_a(cc, voffA0, ahi0, alo0);
    load_a(cc, voffA1, ahi1, alo1);

    // bf16x3 on 4 independent accumulator chains (3 deep each)
    acc00 = WMMA_BF16(ahi0, bh0, acc00);
    acc01 = WMMA_BF16(ahi0, bh1, acc01);
    acc10 = WMMA_BF16(ahi1, bh0, acc10);
    acc11 = WMMA_BF16(ahi1, bh1, acc11);
    acc00 = WMMA_BF16(ahi0, bl0, acc00);
    acc01 = WMMA_BF16(ahi0, bl1, acc01);
    acc10 = WMMA_BF16(ahi1, bl0, acc10);
    acc11 = WMMA_BF16(ahi1, bl1, acc11);
    acc00 = WMMA_BF16(alo0, bh0, acc00);
    acc01 = WMMA_BF16(alo0, bh1, acc01);
    acc10 = WMMA_BF16(alo1, bh0, acc10);
    acc11 = WMMA_BF16(alo1, bh1, acc11);
  };

  // ---- main loop over matched nnz blocks, double-buffered W staging
  if (cnt > 0) { stage_load(__builtin_amdgcn_readfirstlane(s_n[0])); stage_store(0); }
  __syncthreads();
  for (int it = 0; it < cnt; ++it) {
    const bool has_next = (it + 1) < cnt;
    if (has_next) stage_load(__builtin_amdgcn_readfirstlane(s_n[it + 1]));
    compute(it & 1, __builtin_amdgcn_readfirstlane(s_c[it]));
    if (has_next) stage_store((it + 1) & 1);
    __syncthreads();
  }

  // ---- store: C/D layout VGPR i -> M = i + (lane<16?0:8), N = lane&15
  float* yb0 = y + (size_t)(rowbase + (lo16 ? 0 : 8)) * OUTFEAT + rb * BLK + m;
  float* yb1 = yb0 + (size_t)16 * OUTFEAT;
#pragma unroll
  for (int i = 0; i < 8; ++i) {
    yb0[0]  = acc00[i];
    yb0[16] = acc01[i];
    yb1[0]  = acc10[i];
    yb1[16] = acc11[i];
    yb0 += OUTFEAT;
    yb1 += OUTFEAT;
  }
}

extern "C" void kernel_launch(void* const* d_in, const int* in_sizes, int n_in,
                              void* d_out, int out_size, void* d_ws, size_t ws_size,
                              hipStream_t stream) {
  const float* x    = (const float*)d_in[0];
  const float* w    = (const float*)d_in[1];
  const int*   rows = (const int*)d_in[2];
  const int*   cols = (const int*)d_in[3];
  float*       y    = (float*)d_out;
  const int    nnz  = in_sizes[2];

  const size_t xelems = (size_t)BATCH * INFEAT;
  const size_t welems = (size_t)nnz * BLK * BLK;
  const size_t need   = 2 * xelems * 2 + 2 * welems * 2;   // hi+lo bf16 for x and w

  dim3 grid(BATCH / BM, OUTFEAT / BLK);
  dim3 block(NTHREADS);

  if (ws_size >= need) {
    char* p = (char*)d_ws;
    unsigned short* xhi = (unsigned short*)p; p += xelems * 2;
    unsigned short* xlo = (unsigned short*)p; p += xelems * 2;
    unsigned short* whi = (unsigned short*)p; p += welems * 2;
    unsigned short* wlo = (unsigned short*)p;
    const int xn4 = (int)(xelems / 4);
    const int wn4 = (int)(welems / 4);
    cvt_split_kernel<<<(xn4 + 255) / 256, 256, 0, stream>>>(x, xhi, xlo, xn4);
    cvt_split_kernel<<<(wn4 + 255) / 256, 256, 0, stream>>>(w, whi, wlo, wn4);
    bsdsd_kernel<true><<<grid, block, 0, stream>>>(x, w, rows, cols, y, nnz,
                                                   xhi, xlo, whi, wlo);
  } else {
    bsdsd_kernel<false><<<grid, block, 0, stream>>>(x, w, rows, cols, y, nnz,
                                                    nullptr, nullptr, nullptr, nullptr);
  }
}
